// GroupedQueryAttention_39788577030254
// MI455X (gfx1250) — matrix operand order repaired and checked by reading, not compiled, not run on hardware
//
#include <hip/hip_runtime.h>
#include <hip/hip_bf16.h>

typedef _Float16 half_t;
typedef __attribute__((ext_vector_type(16))) _Float16 v16h;
typedef __attribute__((ext_vector_type(8)))  float    v8f;
typedef __attribute__((ext_vector_type(4)))  int      v4i;

#define T_SEQ   2048
#define D_MODEL 512
#define N_HQ    8
#define N_HKV   4
#define D_HEAD  64
#define LOG2E   1.44269504088896340736f

#if defined(__has_builtin)
#if __has_builtin(__builtin_amdgcn_global_load_async_to_lds_b128) && \
    __has_builtin(__builtin_amdgcn_s_wait_asynccnt)
#define USE_ASYNC_LDS 1
#endif
#endif

__device__ inline v8f wmma16(v16h a, v16h b, v8f c) {
  // D = A(16x32 f16) * B(32x16 f16) + C(16x16 f32)
  return __builtin_amdgcn_wmma_f32_16x16x32_f16(false, a, false, b, (short)0, c,
                                                false, false);
}

// 32 contiguous bytes -> one fragment register set (32B aligned)
__device__ inline v16h ld32(const half_t* p) {
  return *reinterpret_cast<const v16h*>(p);
}
// two 16B runs at p and p+8 halves (A-fragment pattern: K = kb..kb+15)
__device__ inline v16h ld2x16(const half_t* p) {
  union U { uint4 u[2]; v16h v; } t;
  t.u[0] = *reinterpret_cast<const uint4*>(p);
  t.u[1] = *reinterpret_cast<const uint4*>(p + 8);
  return t.v;
}

// ---------------- weight preparation (fragment-ordered B matrices) ----------------
// Fragment chunk = 32x16 block of B[k][n] = W[n][k]; chunk index = kb32*(N/16)+nb16;
// element [lane*16 + i] = B[(lane>>4)*16 + i][lane&15] of that block.

__global__ void conv_w_frag(const float* __restrict__ W, half_t* __restrict__ Wf,
                            float scale) {
  int idx = blockIdx.x * 256 + threadIdx.x;   // 512*512 total, N = 512
  int i = idx & 15;
  int lane = (idx >> 4) & 31;
  int chunk = idx >> 9;
  int nb = chunk & 31;                        // N/16 = 32
  int kb = chunk >> 5;
  int k = (kb << 5) + ((lane >> 4) << 4) + i;
  int n = (nb << 4) + (lane & 15);
  Wf[idx] = (half_t)(W[(size_t)n * 512 + k] * scale);
}

// Combined per-head weight (K/V paths): B[k][h*64+dq] = sum_dk W[h*128+dk][k]*proj[dq][dk]
__global__ void comb_w_frag(const float* __restrict__ W,
                            const float* __restrict__ proj,
                            half_t* __restrict__ Wf) {
  int idx = blockIdx.x * 256 + threadIdx.x;   // 512*256 total, N = 256
  int i = idx & 15;
  int lane = (idx >> 4) & 31;
  int chunk = idx >> 9;
  int nb = chunk & 15;                        // N/16 = 16
  int kb = chunk >> 4;
  int k = (kb << 5) + ((lane >> 4) << 4) + i;
  int n = (nb << 4) + (lane & 15);
  int h = n >> 6, dq = n & 63;
  float s = 0.f;
#pragma unroll 4
  for (int dk = 0; dk < 128; ++dk)
    s += W[(size_t)(h * 128 + dk) * 512 + k] * proj[dq * 128 + dk];
  Wf[idx] = (half_t)s;
}

__global__ void prep_bias_kernel(const float* __restrict__ Wq_b,
                                 const float* __restrict__ Wo_b,
                                 const float* __restrict__ Wk_b,
                                 const float* __restrict__ mproj_w,
                                 const float* __restrict__ mproj_b,
                                 const float* __restrict__ Wv_b,
                                 const float* __restrict__ vproj_w,
                                 const float* __restrict__ vproj_b,
                                 float* __restrict__ bq, float* __restrict__ bo,
                                 float* __restrict__ bk, float* __restrict__ bv) {
  int t = threadIdx.x;
  if (blockIdx.x == 0) {
    bq[t] = Wq_b[t] * (0.125f * LOG2E);       // fold 1/sqrt(DQ) and log2(e)
  } else if (blockIdx.x == 1) {
    bo[t] = Wo_b[t];
  } else if (blockIdx.x == 2) {
    if (t < 256) {
      int h = t >> 6, dq = t & 63;
      float s = mproj_b[dq];
      for (int dk = 0; dk < 128; ++dk)
        s += Wk_b[h * 128 + dk] * mproj_w[dq * 128 + dk];
      bk[t] = s;
    }
  } else {
    if (t < 256) {
      int h = t >> 6, dq = t & 63;
      float s = vproj_b[dq];
      for (int dk = 0; dk < 128; ++dk)
        s += Wv_b[h * 128 + dk] * vproj_w[dq * 128 + dk];
      bv[t] = s;
    }
  }
}

// ---------------- projection GEMM (fp32 A -> f16 out, double-buffered A tile) ----------------
// C[M=8192,N] = A @ Wf + bias. LAYOUT 0: head-major [B,H,T,64]; LAYOUT 1: V-fragment order.
template <int LAYOUT>
__global__ __launch_bounds__(128)
void proj_gemm(const float* __restrict__ A, const half_t* __restrict__ Wf,
               const float* __restrict__ bias, half_t* __restrict__ Out,
               int N, int H) {
  __shared__ __align__(16) half_t As[2][128 * 48];
  int tid = threadIdx.x, lane = tid & 31, w = tid >> 5;
  int col0 = lane & 15, hsel = lane >> 4;
  int mBase = blockIdx.x << 7;
  int nBase = blockIdx.y << 6;
  auto loadTile = [&](int kk, half_t* dst0) {
    const float4* src =
        reinterpret_cast<const float4*>(A + (size_t)(mBase + tid) * D_MODEL + kk);
    half_t* dst = dst0 + tid * 48;
#pragma unroll
    for (int i = 0; i < 8; ++i) {
      float4 f = src[i];
      dst[4 * i + 0] = (half_t)f.x;
      dst[4 * i + 1] = (half_t)f.y;
      dst[4 * i + 2] = (half_t)f.z;
      dst[4 * i + 3] = (half_t)f.w;
    }
  };
  v8f acc[2][4] = {};
  loadTile(0, As[0]);
  __syncthreads();
  int p = 0;
  for (int kk = 0; kk < 512; kk += 32) {
    if (kk + 32 < 512) {
      __builtin_prefetch(A + (size_t)(mBase + tid) * D_MODEL + kk + 64, 0, 1);
      loadTile(kk + 32, As[p ^ 1]);
    }
    const half_t* a0p = As[p] + ((w << 5) + col0) * 48 + (hsel << 4);
    v16h a0 = ld2x16(a0p);
    v16h a1 = ld2x16(a0p + 16 * 48);
    const half_t* bp =
        Wf + ((size_t)(kk >> 5) * (N >> 4) + (nBase >> 4)) * 512 + lane * 16;
#pragma unroll
    for (int cb = 0; cb < 4; ++cb) {
      v16h bfr = ld32(bp + (size_t)cb * 512);
      acc[0][cb] = wmma16(a0, bfr, acc[0][cb]);
      acc[1][cb] = wmma16(a1, bfr, acc[1][cb]);
    }
    __syncthreads();
    p ^= 1;
  }
#pragma unroll
  for (int sub = 0; sub < 2; ++sub)
#pragma unroll
    for (int cb = 0; cb < 4; ++cb)
#pragma unroll
      for (int r = 0; r < 8; ++r) {
        int m = mBase + (w << 5) + (sub << 4) + r + (hsel << 3);
        int n = nBase + (cb << 4) + col0;
        int bb = m >> 11, t = m & 2047;
        int head = n >> 6, dq = n & 63;
        float v = acc[sub][cb][r] + bias[n];
        if (LAYOUT == 0) {
          Out[(((size_t)bb * H + head) * T_SEQ + t) * D_HEAD + dq] = (half_t)v;
        } else {
          // V-fragment order: chunk = (bh*(T/32) + key/32)*4 + dq/16
          int kb = t >> 5, hi = (t >> 4) & 1, ii = t & 15;
          size_t chunk =
              ((size_t)(bb * H + head) * (T_SEQ >> 5) + kb) * 4 + (dq >> 4);
          Out[chunk * 512 + (size_t)((hi << 4) + (dq & 15)) * 16 + ii] = (half_t)v;
        }
      }
}

// ---------------- flash attention ----------------
// grid: x = T/64 row tiles, y = B*HQ. block = 128 (4 waves, 16 query rows each).
// Scores are in log2 domain (log2e folded into Wq), so softmax uses raw exp2.
__global__ __launch_bounds__(128)
void attn_kernel(const half_t* __restrict__ qh, const half_t* __restrict__ kh,
                 const half_t* __restrict__ vf, half_t* __restrict__ X) {
  __shared__ __align__(16) half_t Ps[4 * 16 * 80];
  int tid = threadIdx.x, lane = tid & 31, w = tid >> 5;
  int col0 = lane & 15, hsel = lane >> 4;
  int rowTile = blockIdx.x;
  int rowBase = rowTile << 6;
  int bh = blockIdx.y;                  // b*HQ + hq
  int b = bh >> 3, hq = bh & 7;
  int kvh = (b << 2) + (hq & 3);        // hq = g*HKV + h  ->  h = hq & 3
  // Q A-fragments (row = lane&15), reused across all key tiles
  const half_t* qp =
      qh + ((size_t)bh * T_SEQ + rowBase + (w << 4) + col0) * D_HEAD + (hsel << 4);
  v16h qf0 = ld2x16(qp);
  v16h qf1 = ld2x16(qp + 32);
  float rm[8], rl[8];
#pragma unroll
  for (int r = 0; r < 8; ++r) { rm[r] = -1e30f; rl[r] = 0.f; }
  v8f o4[4] = {};
  half_t* Pw = Ps + (w << 4) * 80;
  size_t krow = (size_t)kvh * T_SEQ;
  size_t vbase = (size_t)kvh * (T_SEQ >> 5) * 4;
  for (int kt = 0; kt <= rowTile; ++kt) {
    int keyBase = kt << 6;
    if (kt < rowTile) {   // prefetch next K/V tile
      __builtin_prefetch(kh + (krow + keyBase + 64 + (tid >> 1)) * D_HEAD, 0, 1);
      __builtin_prefetch(
          vf + (vbase + (size_t)(((keyBase >> 5) + 2) * 4)) * 512 + tid * 16, 0, 1);
    }
    // S = Q * K^T : B-fragment of K^T = 16 contiguous halves of one K row
    v8f s4[4] = {};
#pragma unroll
    for (int k0 = 0; k0 < 2; ++k0) {
      v16h a = (k0 == 0) ? qf0 : qf1;
#pragma unroll
      for (int cb = 0; cb < 4; ++cb) {
        const half_t* kp = kh + (krow + keyBase + (cb << 4) + col0) * D_HEAD +
                           (k0 << 5) + (hsel << 4);
        s4[cb] = wmma16(a, ld32(kp), s4[cb]);
      }
    }
    // causal mask + online softmax; row r lives on 16 lanes of one half-wave
    bool diag = (kt == rowTile);
#pragma unroll
    for (int r = 0; r < 8; ++r) {
      int grow = rowBase + (w << 4) + r + (hsel << 3);
      if (diag) {
#pragma unroll
        for (int cb = 0; cb < 4; ++cb) {
          int gcol = keyBase + (cb << 4) + col0;
          if (gcol > grow) s4[cb][r] = -1e30f;
        }
      }
      float mloc = s4[0][r];
#pragma unroll
      for (int cb = 1; cb < 4; ++cb) mloc = fmaxf(mloc, s4[cb][r]);
#pragma unroll
      for (int off = 1; off < 16; off <<= 1)
        mloc = fmaxf(mloc, __shfl_xor(mloc, off, 32));
      float mnew = fmaxf(rm[r], mloc);
      float alpha = exp2f(rm[r] - mnew);
      rm[r] = mnew;
      float ls = 0.f;
#pragma unroll
      for (int cb = 0; cb < 4; ++cb) {
        float pv = exp2f(s4[cb][r] - mnew);
        s4[cb][r] = pv;
        ls += pv;
      }
#pragma unroll
      for (int off = 1; off < 16; off <<= 1) ls += __shfl_xor(ls, off, 32);
      rl[r] = rl[r] * alpha + ls;
#pragma unroll
      for (int cb = 0; cb < 4; ++cb) o4[cb][r] *= alpha;
    }
    // stage P (16x64) into per-wave LDS to re-layout C->A fragments (wave-local, no barrier)
#pragma unroll
    for (int r = 0; r < 8; ++r)
#pragma unroll
      for (int cb = 0; cb < 4; ++cb)
        Pw[(r + (hsel << 3)) * 80 + (cb << 4) + col0] = (half_t)s4[cb][r];
    // O += P * V  (V fragments direct from fragment-ordered global)
#pragma unroll
    for (int k0 = 0; k0 < 2; ++k0) {
      const half_t* pp = Pw + col0 * 80 + (hsel << 4) + (k0 << 5);
      v16h a = ld2x16(pp);
#pragma unroll
      for (int cb = 0; cb < 4; ++cb) {
        const half_t* vp =
            vf + (vbase + (size_t)(((keyBase >> 5) + k0) * 4 + cb)) * 512 + lane * 16;
        o4[cb] = wmma16(a, ld32(vp), o4[cb]);
      }
    }
  }
  // epilogue: O / l  -> X[b, t, hq*64 + d]  (f16)
#pragma unroll
  for (int r = 0; r < 8; ++r) {
    int grow = rowBase + (w << 4) + r + (hsel << 3);
    float inv = 1.f / rl[r];
#pragma unroll
    for (int cb = 0; cb < 4; ++cb) {
      int d = (cb << 4) + col0;
      X[((size_t)b * T_SEQ + grow) * D_MODEL + hq * D_HEAD + d] =
          (half_t)(o4[cb][r] * inv);
    }
  }
}

// ---------------- output GEMM (f16 A -> fp32 out, async LDS copy when available) ----------------
__global__ __launch_bounds__(128)
void out_gemm(const half_t* __restrict__ A, const half_t* __restrict__ Wf,
              const float* __restrict__ bias, float* __restrict__ Out) {
  __shared__ __align__(16) half_t As[2][128 * 48];
  int tid = threadIdx.x, lane = tid & 31, w = tid >> 5;
  int col0 = lane & 15, hsel = lane >> 4;
  int mBase = blockIdx.x << 7;
  int nBase = blockIdx.y << 6;
  auto loadTile = [&](int kk, half_t* dst0) {
    const half_t* src = A + (size_t)(mBase + tid) * D_MODEL + kk;
    half_t* dst = dst0 + tid * 48;
#ifdef USE_ASYNC_LDS
#pragma unroll
    for (int i = 0; i < 4; ++i)
      __builtin_amdgcn_global_load_async_to_lds_b128(
          (v4i*)(src + i * 8), (v4i*)(dst + i * 8), 0, 0);
#else
    const uint4* s4 = reinterpret_cast<const uint4*>(src);
    uint4* d4 = reinterpret_cast<uint4*>(dst);
#pragma unroll
    for (int i = 0; i < 4; ++i) d4[i] = s4[i];
#endif
  };
  v8f acc[2][4] = {};
  loadTile(0, As[0]);
#ifdef USE_ASYNC_LDS
  __builtin_amdgcn_s_wait_asynccnt(0);
#endif
  __syncthreads();
  int p = 0;
  for (int kk = 0; kk < 512; kk += 32) {
    if (kk + 32 < 512) {
      __builtin_prefetch(A + (size_t)(mBase + tid) * D_MODEL + kk + 64, 0, 1);
      loadTile(kk + 32, As[p ^ 1]);
    }
    const half_t* a0p = As[p] + ((w << 5) + col0) * 48 + (hsel << 4);
    v16h a0 = ld2x16(a0p);
    v16h a1 = ld2x16(a0p + 16 * 48);
    const half_t* bp =
        Wf + ((size_t)(kk >> 5) * 32 + (nBase >> 4)) * 512 + lane * 16;
#pragma unroll
    for (int cb = 0; cb < 4; ++cb) {
      v16h bfr = ld32(bp + (size_t)cb * 512);
      acc[0][cb] = wmma16(a0, bfr, acc[0][cb]);
      acc[1][cb] = wmma16(a1, bfr, acc[1][cb]);
    }
#ifdef USE_ASYNC_LDS
    __builtin_amdgcn_s_wait_asynccnt(0);
#endif
    __syncthreads();
    p ^= 1;
  }
#pragma unroll
  for (int sub = 0; sub < 2; ++sub)
#pragma unroll
    for (int cb = 0; cb < 4; ++cb)
#pragma unroll
      for (int r = 0; r < 8; ++r) {
        int m = mBase + (w << 5) + (sub << 4) + r + (hsel << 3);
        int n = nBase + (cb << 4) + col0;
        Out[(size_t)m * D_MODEL + n] = acc[sub][cb][r] + bias[n];
      }
}

extern "C" void kernel_launch(void* const* d_in, const int* in_sizes, int n_in,
                              void* d_out, int out_size, void* d_ws, size_t ws_size,
                              hipStream_t stream) {
  (void)in_sizes; (void)n_in; (void)out_size; (void)ws_size;
  const float* Q       = (const float*)d_in[0];
  const float* K       = (const float*)d_in[1];
  const float* V       = (const float*)d_in[2];
  const float* Wq_w    = (const float*)d_in[3];
  const float* Wq_b    = (const float*)d_in[4];
  const float* Wk_w    = (const float*)d_in[5];
  const float* Wk_b    = (const float*)d_in[6];
  const float* Wv_w    = (const float*)d_in[7];
  const float* Wv_b    = (const float*)d_in[8];
  const float* Wo_w    = (const float*)d_in[9];
  const float* Wo_b    = (const float*)d_in[10];
  const float* vproj_w = (const float*)d_in[11];
  const float* vproj_b = (const float*)d_in[12];
  const float* mproj_w = (const float*)d_in[13];
  const float* mproj_b = (const float*)d_in[14];

  char* ws = (char*)d_ws;
  size_t off = 0;
  auto alloc = [&](size_t bytes) {
    char* p = ws + off;
    off = (off + bytes + 255) & ~(size_t)255;
    return p;
  };
  half_t* wtq = (half_t*)alloc((size_t)512 * 512 * 2);
  half_t* wto = (half_t*)alloc((size_t)512 * 512 * 2);
  half_t* wtk = (half_t*)alloc((size_t)512 * 256 * 2);
  half_t* wtv = (half_t*)alloc((size_t)512 * 256 * 2);
  float*  bq  = (float*)alloc(512 * 4);
  float*  bo  = (float*)alloc(512 * 4);
  float*  bk  = (float*)alloc(256 * 4);
  float*  bv  = (float*)alloc(256 * 4);
  half_t* qhw = (half_t*)alloc((size_t)4 * N_HQ  * T_SEQ * D_HEAD * 2);
  half_t* khw = (half_t*)alloc((size_t)4 * N_HKV * T_SEQ * D_HEAD * 2);
  half_t* vfw = (half_t*)alloc((size_t)4 * N_HKV * T_SEQ * D_HEAD * 2);
  half_t* Xa  = (half_t*)alloc((size_t)4 * T_SEQ * D_MODEL * 2);

  conv_w_frag<<<1024, 256, 0, stream>>>(Wq_w, wtq, 0.125f * LOG2E);
  conv_w_frag<<<1024, 256, 0, stream>>>(Wo_w, wto, 1.0f);
  comb_w_frag<<<512, 256, 0, stream>>>(Wk_w, mproj_w, wtk);
  comb_w_frag<<<512, 256, 0, stream>>>(Wv_w, vproj_w, wtv);
  prep_bias_kernel<<<4, 512, 0, stream>>>(Wq_b, Wo_b, Wk_b, mproj_w, mproj_b,
                                          Wv_b, vproj_w, vproj_b, bq, bo, bk, bv);
  proj_gemm<0><<<dim3(64, 8), 128, 0, stream>>>(Q, wtq, bq, qhw, 512, N_HQ);
  proj_gemm<0><<<dim3(64, 4), 128, 0, stream>>>(K, wtk, bk, khw, 256, N_HKV);
  proj_gemm<1><<<dim3(64, 4), 128, 0, stream>>>(V, wtv, bv, vfw, 256, N_HKV);
  attn_kernel<<<dim3(32, 32), 128, 0, stream>>>(qhw, khw, vfw, Xa);
  out_gemm<<<dim3(64, 8), 128, 0, stream>>>(Xa, wto, bo, (float*)d_out);
}
